// SemanticGraphAggregator_41111426957461
// MI455X (gfx1250) — compile-verified, matrix operand order checked by reading, not run on hardware
//
#include <hip/hip_runtime.h>
#include <hip/hip_bf16.h>

// ---------------------------------------------------------------------------
// CDNA5 (gfx1250) wave32 WMMA implementation of the SemanticGraphAggregator.
// bf16 WMMA (v_wmma_f32_16x16x32_bf16) everywhere, f32 accumulation.
// Async global->LDS staging (ASYNCcnt) when the toolchain exposes it.
// ---------------------------------------------------------------------------

typedef __attribute__((ext_vector_type(16))) __bf16 bf16x16;
typedef __attribute__((ext_vector_type(8)))  float  floatx8;
typedef __attribute__((ext_vector_type(4)))  int    int4v;

#ifndef __has_builtin
#define __has_builtin(x) 0
#endif
#if __has_builtin(__builtin_amdgcn_global_load_async_to_lds_b128) && \
    __has_builtin(__builtin_amdgcn_s_wait_asynccnt)
#define CDNA5_ASYNC_LDS 1
#else
#define CDNA5_ASYNC_LDS 0
#endif

#if CDNA5_ASYNC_LDS
typedef __attribute__((address_space(1))) int4v* gv4ptr;  // global src
typedef __attribute__((address_space(3))) int4v* lv4ptr;  // LDS dst
#endif

// 16-byte global -> LDS copy; async (GLOBAL_LOAD_ASYNC_TO_LDS_B128) if available.
// AS casts go through integers: generic->AS1 is value-identical; the low 32
// bits of a generic LDS address are the LDS byte offset (ISA 10.2 aperture map).
__device__ __forceinline__ void copy16_g2l(const void* g, void* l) {
#if CDNA5_ASYNC_LDS
  __builtin_amdgcn_global_load_async_to_lds_b128(
      (gv4ptr)(size_t)g, (lv4ptr)(unsigned)(size_t)l, 0, 0);
#else
  *(uint4*)l = *(const uint4*)g;
#endif
}
__device__ __forceinline__ void copy_wait() {
#if CDNA5_ASYNC_LDS
  __builtin_amdgcn_s_wait_asynccnt(0);
#endif
}

__device__ __forceinline__ floatx8 wmma_bf16(bf16x16 a, bf16x16 b, floatx8 c) {
  // 8 args: (neg_a, A, neg_b, B, c_mod, C, reuse_a, reuse_b)
  return __builtin_amdgcn_wmma_f32_16x16x32_bf16(false, a, false, b, (short)0, c,
                                                 false, false);
}

// 16x32 fragment, source stored row-major: elem[row][k] at base[row*stride + k].
// Used for A operands and for B operands given as B^T (weights [N,K]).
// ISA 7.12.2: lane&15 = row, lane>>4 selects K half (+8), VGPR v holds pair
// {2v,2v+1} (v<4) or {16+2(v-4),...} (v>=4).
__device__ __forceinline__ bf16x16 frag_rowmajor(const __bf16* base, int stride,
                                                 int lane) {
  union { bf16x16 v; unsigned u[8]; } f;
  const int m = lane & 15, hl = lane >> 4;
  const __bf16* row = base + (size_t)m * stride + 8 * hl;
  #pragma unroll
  for (int i = 0; i < 8; ++i) {
    const int k = (i & 4) ? (16 + 2 * (i & 3)) : (2 * i);
    f.u[i] = *(const unsigned*)(row + k);  // stride/base even -> 4B aligned
  }
  return f.v;
}

// B fragment when B is stored K-major: elem[k][n] at base[k*stride + n].
__device__ __forceinline__ bf16x16 frag_kmajor(const __bf16* base, int stride,
                                               int lane) {
  union { bf16x16 v; __bf16 h[16]; } f;
  const int n = lane & 15, hl = lane >> 4;
  const __bf16* col = base + n + (size_t)(8 * hl) * stride;
  #pragma unroll
  for (int i = 0; i < 8; ++i) {
    const int k = (i & 4) ? (16 + 2 * (i & 3)) : (2 * i);
    f.h[2 * i]     = col[(size_t)k * stride];
    f.h[2 * i + 1] = col[(size_t)(k + 1) * stride];
  }
  return f.v;
}

// ---------------------------------------------------------------------------
// Weight-norm prep: out[o,:] = bf16( v[o,:] * g[o] / ||v[o,:]|| ), padded to
// Kpad cols / grid rows (zero fill). Column split remap aligns concat padding.
// ---------------------------------------------------------------------------
__global__ void wn_prep_kernel(const float* __restrict__ v,
                               const float* __restrict__ g,
                               __bf16* __restrict__ out,
                               int O, int I, int Kpad, int split_at, int split_pad) {
  const int o = blockIdx.x;
  const int tid = threadIdx.x;
  __shared__ float red[256];
  if (o >= O) {
    for (int c = tid; c < Kpad; c += 256) out[(size_t)o * Kpad + c] = (__bf16)0.f;
    return;
  }
  const float* row = v + (size_t)o * I;
  float ss = 0.f;
  for (int c = tid; c < I; c += 256) { float x = row[c]; ss += x * x; }
  red[tid] = ss;
  __syncthreads();
  for (int s = 128; s > 0; s >>= 1) {
    if (tid < s) red[tid] += red[tid + s];
    __syncthreads();
  }
  const float scale = g[o] / sqrtf(red[0]);
  for (int d = tid; d < Kpad; d += 256) {
    int src; bool valid;
    if (d < split_at)                  { src = d;              valid = (src < I); }
    else if (d < split_at + split_pad) { src = 0;              valid = false;     }
    else                               { src = d - split_pad;  valid = (src < I); }
    out[(size_t)o * Kpad + d] = valid ? (__bf16)(row[src] * scale) : (__bf16)0.f;
  }
}

// s (f32 [51200,300]) -> bf16 padded into cat_ss[:, 0:320] (stride 640)
__global__ void s_convert_kernel(const float* __restrict__ s,
                                 __bf16* __restrict__ cat_ss) {
  size_t idx = (size_t)blockIdx.x * 256 + threadIdx.x;
  if (idx >= (size_t)51200 * 320) return;
  int r = (int)(idx / 320), c = (int)(idx % 320);
  cat_ss[(size_t)r * 640 + c] = (c < 300) ? (__bf16)s[(size_t)r * 300 + c]
                                          : (__bf16)0.f;
}

// out = s (residual base; h_is / h_ss GEMMs accumulate on top)
__global__ void out_init_kernel(const float* __restrict__ s,
                                float* __restrict__ out) {
  size_t idx = (size_t)blockIdx.x * 256 + threadIdx.x;
  if (idx < (size_t)51200 * 300) out[idx] = s[idx];
}

// ---------------------------------------------------------------------------
// Flat GEMM: out = relu(A * W^T + bias). Each block owns 64 rows and the FULL
// N = NTILES*16 columns, so A streams from HBM exactly once and W (<2MB) stays
// L2-resident. 8 waves: wave w -> m-tiles {2*(w&1), +1}, n-tiles
// (w>>1)*NPW..+NPW-1. A-fragments hoisted across the n loop.
// ---------------------------------------------------------------------------
template <typename TA, bool ACCUM, int NTILES>
__global__ void gemm_wn_kernel(const TA* __restrict__ A, int lda,
                               const __bf16* __restrict__ W, int K,
                               const float* __restrict__ bias,
                               void* __restrict__ outp, int ldo, int Nreal) {
  constexpr int NPW = NTILES / 4;  // n-tiles per wave
  __shared__ __bf16 sA[64 * 40];
  __shared__ __bf16 sB[NTILES * 16 * 40];
  const int tid = threadIdx.x, lane = tid & 31, w = tid >> 5;
  const int mp = w & 1, q = w >> 1;
  const int hl = lane >> 4, nl = lane & 15;
  const size_t row0 = (size_t)blockIdx.x * 64;

  floatx8 acc[2][NPW] = {};
  for (int kk = 0; kk < K; kk += 32) {
    __syncthreads();
    // stage B slice: all NTILES*16 weight rows, cols [kk, kk+32)
    for (int t = tid; t < NTILES * 16 * 4; t += 256) {
      const int r = t >> 2, c = (t & 3) * 8;
      copy16_g2l(W + (size_t)r * K + kk + c, sB + r * 40 + c);
    }
    // stage A slice (f32 -> bf16 convert, or bf16 async copy)
    if constexpr (sizeof(TA) == 4) {
      for (int t = tid; t < 64 * 8; t += 256) {
        const int r = t >> 3, c = (t & 7) * 4;
        const float4 v4 = *(const float4*)(A + (row0 + r) * (size_t)lda + kk + c);
        union { __bf16 h[4]; uint2 u; } pk;
        pk.h[0] = (__bf16)v4.x; pk.h[1] = (__bf16)v4.y;
        pk.h[2] = (__bf16)v4.z; pk.h[3] = (__bf16)v4.w;
        *(uint2*)(sA + r * 40 + c) = pk.u;
      }
    } else {
      for (int t = tid; t < 64 * 4; t += 256) {
        const int r = t >> 2, c = (t & 3) * 8;
        copy16_g2l(A + (row0 + r) * (size_t)lda + kk + c, sA + r * 40 + c);
      }
    }
    copy_wait();
    __syncthreads();
    const bf16x16 a0 = frag_rowmajor(sA + (2 * mp + 0) * 16 * 40, 40, lane);
    const bf16x16 a1 = frag_rowmajor(sA + (2 * mp + 1) * 16 * 40, 40, lane);
    #pragma unroll
    for (int j = 0; j < NPW; ++j) {
      const bf16x16 bf = frag_rowmajor(sB + (q * NPW + j) * 16 * 40, 40, lane);
      acc[0][j] = wmma_bf16(a0, bf, acc[0][j]);
      acc[1][j] = wmma_bf16(a1, bf, acc[1][j]);
    }
  }

  #pragma unroll
  for (int j = 0; j < NPW; ++j) {
    const int col = (q * NPW + j) * 16 + nl;
    const float bv = (col < Nreal) ? bias[col] : 0.f;
    #pragma unroll
    for (int p = 0; p < 2; ++p) {
      #pragma unroll
      for (int r = 0; r < 8; ++r) {
        const size_t rowg = row0 + (2 * mp + p) * 16 + r + 8 * hl;
        const float val = fmaxf(acc[p][j][r] + bv, 0.f);
        if (ACCUM) {
          if (col < Nreal) ((float*)outp)[rowg * ldo + col] += val;
        } else {
          if (col < Nreal) ((__bf16*)outp)[rowg * ldo + col] = (__bf16)val;
        }
      }
    }
  }
}

// ---------------------------------------------------------------------------
// Per-batch image->semantic attention, fully in LDS.
// att = softmax(s_proj (50x512) x i_proj^T); i_att = att x i_proj.
// Writes i_att bf16 into cat_is[:, 512:1024].
// ---------------------------------------------------------------------------
__global__ void att_is_kernel(__bf16* __restrict__ cat_is,
                              const __bf16* __restrict__ i_proj) {
  extern __shared__ char smem[];
  __bf16* sS = (__bf16*)smem;                  // 64 x 520  (s_proj, padded)
  __bf16* sI = sS + 64 * 520;                  // 128 x 520 (i_proj, padded)
  float*  sF = (float*)(sI + 128 * 520);       // 64 x 128  (logits)
  __bf16* sT = (__bf16*)(sF + 64 * 128);       // 64 x 128  (att bf16)
  const int b = blockIdx.x, tid = threadIdx.x, lane = tid & 31, w = tid >> 5;
  const int mt = w & 3, grp = w >> 2;
  const int hl = lane >> 4, nl = lane & 15;

  // zero pad rows, then async-stage the valid rows (16B granules)
  for (int t = tid; t < 14 * 520; t += 256) sS[50 * 520 + t] = (__bf16)0.f;
  for (int t = tid; t < 28 * 520; t += 256) sI[100 * 520 + t] = (__bf16)0.f;
  for (int t = tid; t < 50 * 64; t += 256) {
    const int r = t >> 6, c = (t & 63) * 8;
    copy16_g2l(cat_is + ((size_t)b * 50 + r) * 1024 + c, sS + r * 520 + c);
  }
  for (int t = tid; t < 100 * 64; t += 256) {
    const int r = t >> 6, c = (t & 63) * 8;
    copy16_g2l(i_proj + ((size_t)b * 100 + r) * 512 + c, sI + r * 520 + c);
  }
  copy_wait();
  __syncthreads();

  { // logits: (64x512) x (512x128), B = i_proj^T (BT fragments from sI rows)
    floatx8 acc[4] = {};
    for (int kk = 0; kk < 512; kk += 32) {
      const bf16x16 a = frag_rowmajor(sS + mt * 16 * 520 + kk, 520, lane);
      #pragma unroll
      for (int j = 0; j < 4; ++j)
        acc[j] = wmma_bf16(a, frag_rowmajor(sI + (grp * 4 + j) * 16 * 520 + kk,
                                            520, lane), acc[j]);
    }
    #pragma unroll
    for (int j = 0; j < 4; ++j) {
      const int col = (grp * 4 + j) * 16 + nl;
      #pragma unroll
      for (int r = 0; r < 8; ++r)
        sF[(mt * 16 + r + 8 * hl) * 128 + col] = acc[j][r];
    }
  }
  __syncthreads();

  if (tid < 64) { // softmax over 100 image slots
    const int r = tid;
    if (r < 50) {
      float mx = -1e30f;
      for (int c = 0; c < 100; ++c) mx = fmaxf(mx, sF[r * 128 + c]);
      float sm = 0.f;
      for (int c = 0; c < 100; ++c) sm += __expf(sF[r * 128 + c] - mx);
      const float inv = 1.f / sm;
      for (int c = 0; c < 128; ++c)
        sT[r * 128 + c] = (c < 100) ? (__bf16)(__expf(sF[r * 128 + c] - mx) * inv)
                                    : (__bf16)0.f;
    } else {
      for (int c = 0; c < 128; ++c) sT[r * 128 + c] = (__bf16)0.f;
    }
  }
  __syncthreads();

  // i_att = att (64x128) x i_proj (128x512); B is K-major in sI.
  for (int ch = 0; ch < 4; ++ch) {
    floatx8 acc[4] = {};
    for (int kk = 0; kk < 128; kk += 32) {
      const bf16x16 a = frag_rowmajor(sT + mt * 16 * 128 + kk, 128, lane);
      #pragma unroll
      for (int j = 0; j < 4; ++j) {
        const int nt = grp * 16 + ch * 4 + j;
        acc[j] = wmma_bf16(a, frag_kmajor(sI + (size_t)kk * 520 + nt * 16,
                                          520, lane), acc[j]);
      }
    }
    #pragma unroll
    for (int j = 0; j < 4; ++j) {
      const int col = (grp * 16 + ch * 4 + j) * 16 + nl;
      #pragma unroll
      for (int r = 0; r < 8; ++r) {
        const int m = mt * 16 + r + 8 * hl;
        if (m < 50)
          cat_is[((size_t)b * 50 + m) * 1024 + 512 + col] = (__bf16)acc[j][r];
      }
    }
  }
}

// ---------------------------------------------------------------------------
// Per-batch semantic self-attention (fused p1/p2 projections + masked softmax).
// Writes s_att bf16 into cat_ss[:, 320:640].
// ---------------------------------------------------------------------------
__global__ void att_ss_kernel(__bf16* __restrict__ cat_ss,
                              const __bf16* __restrict__ W4,
                              const __bf16* __restrict__ W5,
                              const float* __restrict__ b4,
                              const float* __restrict__ b5,
                              const int* __restrict__ mask_s) {
  extern __shared__ char smem[];
  __bf16* sX  = (__bf16*)smem;                 // 64 x 328 (s bf16, padded)
  __bf16* sP1 = sX + 64 * 328;                 // 64 x 520
  __bf16* sP2 = sP1 + 64 * 520;                // 64 x 520
  float*  sF  = (float*)(sP2 + 64 * 520);      // 64 x 64 logits
  __bf16* sT  = (__bf16*)(sF + 64 * 64);       // 64 x 64 att2
  const int b = blockIdx.x, tid = threadIdx.x, lane = tid & 31, w = tid >> 5;
  const int mt = w & 3, grp = w >> 2;
  const int hl = lane >> 4, nl = lane & 15;

  for (int t = tid; t < 14 * 328; t += 256) sX[50 * 328 + t] = (__bf16)0.f;
  for (int t = tid; t < 50 * 40; t += 256) {
    const int r = t / 40, c = (t % 40) * 8;
    copy16_g2l(cat_ss + ((size_t)b * 50 + r) * 640 + c, sX + r * 328 + c);
  }
  copy_wait();
  __syncthreads();

  // p1 = relu(sX x W4^T + b4), p2 likewise (M=64, N=512, K=320)
  #pragma unroll 1
  for (int which = 0; which < 2; ++which) {
    const __bf16* Wg = which ? W5 : W4;
    const float*  bg = which ? b5 : b4;
    __bf16* dst = which ? sP2 : sP1;
    for (int ch = 0; ch < 4; ++ch) {
      floatx8 acc[4] = {};
      for (int kk = 0; kk < 320; kk += 32) {
        const bf16x16 a = frag_rowmajor(sX + mt * 16 * 328 + kk, 328, lane);
        #pragma unroll
        for (int j = 0; j < 4; ++j) {
          const int nt = grp * 16 + ch * 4 + j;
          acc[j] = wmma_bf16(a, frag_rowmajor(Wg + (size_t)nt * 16 * 320 + kk,
                                              320, lane), acc[j]);
        }
      }
      #pragma unroll
      for (int j = 0; j < 4; ++j) {
        const int col = (grp * 16 + ch * 4 + j) * 16 + nl;
        #pragma unroll
        for (int r = 0; r < 8; ++r) {
          const int m = mt * 16 + r + 8 * hl;
          dst[m * 520 + col] = (__bf16)fmaxf(acc[j][r] + bg[col], 0.f);
        }
      }
    }
  }
  __syncthreads();

  { // sim = p1 x p2^T  (64x64, K=512)
    floatx8 acc[2] = {};
    for (int kk = 0; kk < 512; kk += 32) {
      const bf16x16 a = frag_rowmajor(sP1 + mt * 16 * 520 + kk, 520, lane);
      #pragma unroll
      for (int j = 0; j < 2; ++j)
        acc[j] = wmma_bf16(a, frag_rowmajor(sP2 + (2 * grp + j) * 16 * 520 + kk,
                                            520, lane), acc[j]);
    }
    #pragma unroll
    for (int j = 0; j < 2; ++j) {
      const int col = (2 * grp + j) * 16 + nl;
      #pragma unroll
      for (int r = 0; r < 8; ++r)
        sF[(mt * 16 + r + 8 * hl) * 64 + col] = acc[j][r];
    }
  }
  __syncthreads();

  { // masked softmax over keys t < mask_s[b]
    int nv = mask_s[b];
    nv = nv < 1 ? 1 : (nv > 50 ? 50 : nv);
    if (tid < 64) {
      const int r = tid;
      if (r < 50) {
        float mx = -1e30f;
        for (int c = 0; c < nv; ++c) mx = fmaxf(mx, sF[r * 64 + c]);
        float sm = 0.f;
        for (int c = 0; c < nv; ++c) sm += __expf(sF[r * 64 + c] - mx);
        const float inv = 1.f / sm;
        for (int c = 0; c < 64; ++c)
          sT[r * 64 + c] = (c < nv) ? (__bf16)(__expf(sF[r * 64 + c] - mx) * inv)
                                    : (__bf16)0.f;
      } else {
        for (int c = 0; c < 64; ++c) sT[r * 64 + c] = (__bf16)0.f;
      }
    }
  }
  __syncthreads();

  // s_att = att2 (64x64) x sX (64x320); B K-major in sX.
  for (int ch = 0; ch < 2; ++ch) {
    floatx8 acc[5] = {};
    for (int kk = 0; kk < 64; kk += 32) {
      const bf16x16 a = frag_rowmajor(sT + mt * 16 * 64 + kk, 64, lane);
      #pragma unroll
      for (int j = 0; j < 5; ++j) {
        const int nt = grp * 10 + ch * 5 + j;
        acc[j] = wmma_bf16(a, frag_kmajor(sX + (size_t)kk * 328 + nt * 16,
                                          328, lane), acc[j]);
      }
    }
    #pragma unroll
    for (int j = 0; j < 5; ++j) {
      const int col = (grp * 10 + ch * 5 + j) * 16 + nl;
      #pragma unroll
      for (int r = 0; r < 8; ++r) {
        const int m = mt * 16 + r + 8 * hl;
        if (m < 50)
          cat_ss[((size_t)b * 50 + m) * 640 + 320 + col] = (__bf16)acc[j][r];
      }
    }
  }
}

// ---------------------------------------------------------------------------
extern "C" void kernel_launch(void* const* d_in, const int* in_sizes, int n_in,
                              void* d_out, int out_size, void* d_ws, size_t ws_size,
                              hipStream_t stream) {
  (void)in_sizes; (void)n_in; (void)out_size; (void)ws_size;
  const float* i_in = (const float*)d_in[0];
  const float* s_in = (const float*)d_in[1];
  const int*   mask = (const int*)d_in[2];
  const float* imv = (const float*)d_in[3],  *img = (const float*)d_in[4],  *imb = (const float*)d_in[5];
  const float* sev = (const float*)d_in[6],  *seg = (const float*)d_in[7],  *seb = (const float*)d_in[8];
  const float* icv = (const float*)d_in[9],  *icg = (const float*)d_in[10], *icb = (const float*)d_in[11];
  const float* s1v = (const float*)d_in[12], *s1g = (const float*)d_in[13], *s1b = (const float*)d_in[14];
  const float* s2v = (const float*)d_in[15], *s2g = (const float*)d_in[16], *s2b = (const float*)d_in[17];
  const float* scv = (const float*)d_in[18], *scg = (const float*)d_in[19], *scb = (const float*)d_in[20];

  char* ws = (char*)d_ws;
  size_t off = 0;
  auto alloc = [&](size_t bytes) -> void* {
    void* p = ws + off;
    off += (bytes + 255) & ~(size_t)255;
    return p;
  };
  __bf16* W1 = (__bf16*)alloc((size_t)512 * 2048 * 2);   // im_is  [512,2048]
  __bf16* W2 = (__bf16*)alloc((size_t)512 * 320 * 2);    // sem_is [512,320]
  __bf16* W3 = (__bf16*)alloc((size_t)320 * 1024 * 2);   // isc    [320,1024]
  __bf16* W4 = (__bf16*)alloc((size_t)512 * 320 * 2);    // ss1
  __bf16* W5 = (__bf16*)alloc((size_t)512 * 320 * 2);    // ss2
  __bf16* W6 = (__bf16*)alloc((size_t)320 * 640 * 2);    // ssc (split 300|+20)
  __bf16* iproj = (__bf16*)alloc((size_t)102400 * 512 * 2);
  __bf16* catis = (__bf16*)alloc((size_t)51200 * 1024 * 2); // [s_proj | i_att]
  __bf16* catss = (__bf16*)alloc((size_t)51200 * 640 * 2);  // [s_bf   | s_att]

  const int NOSPLIT = 1 << 30;
  wn_prep_kernel<<<512, 256, 0, stream>>>(imv, img, W1, 512, 2048, 2048, NOSPLIT, 0);
  wn_prep_kernel<<<512, 256, 0, stream>>>(sev, seg, W2, 512, 300, 320, NOSPLIT, 0);
  wn_prep_kernel<<<320, 256, 0, stream>>>(icv, icg, W3, 300, 1024, 1024, NOSPLIT, 0);
  wn_prep_kernel<<<512, 256, 0, stream>>>(s1v, s1g, W4, 512, 300, 320, NOSPLIT, 0);
  wn_prep_kernel<<<512, 256, 0, stream>>>(s2v, s2g, W5, 512, 300, 320, NOSPLIT, 0);
  wn_prep_kernel<<<320, 256, 0, stream>>>(scv, scg, W6, 300, 600, 640, 300, 20);

  s_convert_kernel<<<(51200 * 320 + 255) / 256, 256, 0, stream>>>(s_in, catss);
  out_init_kernel<<<(51200 * 300 + 255) / 256, 256, 0, stream>>>(s_in, (float*)d_out);

  // i_proj = relu(i x W1^T + b): [102400,2048] x [2048,512] -> bf16 (A read once)
  gemm_wn_kernel<float, false, 32><<<1600, 256, 0, stream>>>(
      i_in, 2048, W1, 2048, imb, iproj, 512, 512);
  // s_proj = relu(s x W2^T + b): [51200,320] x [320,512] -> catis[:,0:512]
  gemm_wn_kernel<__bf16, false, 32><<<800, 256, 0, stream>>>(
      catss, 640, W2, 320, seb, catis, 1024, 512);

  const size_t lds_is =
      ((size_t)64 * 520 + 128 * 520) * 2 + 64 * 128 * 4 + 64 * 128 * 2;
  att_is_kernel<<<1024, 256, lds_is, stream>>>(catis, iproj);

  // out += relu(concat(s_proj, i_att) x W3^T + b)
  gemm_wn_kernel<__bf16, true, 20><<<800, 256, 0, stream>>>(
      catis, 1024, W3, 1024, icb, d_out, 300, 300);

  const size_t lds_ss =
      ((size_t)64 * 328 + 64 * 520 + 64 * 520) * 2 + 64 * 64 * 4 + 64 * 64 * 2;
  att_ss_kernel<<<1024, 256, lds_ss, stream>>>(catss, W4, W5, s1b, s2b, mask);

  // out += relu(concat(s, s_att) x W6^T + b)
  gemm_wn_kernel<__bf16, true, 20><<<800, 256, 0, stream>>>(
      catss, 640, W6, 640, scb, d_out, 300, 300);
}
